// ProbAttention_429496730064
// MI455X (gfx1250) — compile-verified, matrix operand order checked by reading, not run on hardware
//
#include <hip/hip_runtime.h>

#define BN 8
#define HN 8
#define LQ 2048
#define DD 64
#define SK 40
#define NTOP 40
#define NTOP_PAD 64   // 4 WMMA tiles of 16 rows (rows 40..63 are padding)

typedef __attribute__((ext_vector_type(16))) _Float16 v16h;
typedef __attribute__((ext_vector_type(8)))  float    v8f;

// DS_SWIZZLE_B32 group-of-32 butterfly: offset = xor<<10 | or<<5 | and
// (and=0x1f keeps all lanes participating; one DS op, no address setup)
#define SWZ_XOR(v, m) \
  __int_as_float(__builtin_amdgcn_ds_swizzle(__float_as_int(v), (0x1f | ((m) << 10))))

__device__ inline float redmax16(float v) {   // butterfly within each 16-lane half
  v = fmaxf(v, SWZ_XOR(v, 1));
  v = fmaxf(v, SWZ_XOR(v, 2));
  v = fmaxf(v, SWZ_XOR(v, 4));
  v = fmaxf(v, SWZ_XOR(v, 8));
  return v;
}
__device__ inline float redsum16(float v) {
  v += SWZ_XOR(v, 1);
  v += SWZ_XOR(v, 2);
  v += SWZ_XOR(v, 4);
  v += SWZ_XOR(v, 8);
  return v;
}
__device__ inline float redsum32(float v) {   // full-wave32 butterfly
  v += SWZ_XOR(v, 1);
  v += SWZ_XOR(v, 2);
  v += SWZ_XOR(v, 4);
  v += SWZ_XOR(v, 8);
  v += SWZ_XOR(v, 16);
  return v;
}

// ---------------------------------------------------------------------------
// Kernel A: sparsity measure M[bh][q] = max_j(q.k_j) - sum_j(q.k_j)/L
// one wave32 per query; gathered K rows are L2-resident (33.5MB << 192MB L2)
// ---------------------------------------------------------------------------
__global__ __launch_bounds__(256) void mscore_kernel(
    const float* __restrict__ Q, const float* __restrict__ K,
    const int* __restrict__ idx, float* __restrict__ M) {
  int wid  = threadIdx.x >> 5;
  int lane = threadIdx.x & 31;
  int slot = blockIdx.x * 8 + wid;     // == bh*LQ + q (contiguous)
  int q    = slot & (LQ - 1);
  int bh   = slot >> 11;

  const float* qp = Q + (size_t)slot * DD;
  float q0 = qp[lane];
  float q1 = qp[lane + 32];

  const float* kb = K + (size_t)bh * LQ * DD;
  float mx = -1e30f, sm = 0.0f;
  for (int j = 0; j < SK; ++j) {
    int kk = idx[q * SK + j];
    const float* kp = kb + (size_t)kk * DD;
    if (j + 1 < SK)   // warm L2->L0 for the next gathered row
      __builtin_prefetch(kb + (size_t)idx[q * SK + j + 1] * DD + lane, 0, 0);
    float d = redsum32(q0 * kp[lane] + q1 * kp[lane + 32]);
    mx = fmaxf(mx, d);
    sm += d;
  }
  if (lane == 0) M[slot] = mx - sm * (1.0f / LQ);
}

// ---------------------------------------------------------------------------
// Kernel B: iterative top-40 per (b,h) over 2048 M values held in LDS
// ---------------------------------------------------------------------------
__global__ __launch_bounds__(256) void topk_kernel(
    const float* __restrict__ M, int* __restrict__ Mtop) {
  __shared__ float vals[LQ];
  __shared__ float rv[256];
  __shared__ int   ri[256];
  int bh = blockIdx.x;
  int t  = threadIdx.x;
  for (int i = t; i < LQ; i += 256) vals[i] = M[(size_t)bh * LQ + i];
  if (t >= NTOP && t < NTOP_PAD) Mtop[bh * NTOP_PAD + t] = 0;  // padding rows
  __syncthreads();
  for (int it = 0; it < NTOP; ++it) {
    float bv = -1e30f; int bi = 0;
    for (int i = t; i < LQ; i += 256) {
      float v = vals[i];
      if (v > bv || (v == bv && i < bi)) { bv = v; bi = i; }
    }
    rv[t] = bv; ri[t] = bi;
    __syncthreads();
    for (int s = 128; s > 0; s >>= 1) {
      if (t < s) {
        float ov = rv[t + s]; int oi = ri[t + s];
        if (ov > rv[t] || (ov == rv[t] && oi < ri[t])) { rv[t] = ov; ri[t] = oi; }
      }
      __syncthreads();
    }
    if (t == 0) { Mtop[bh * NTOP_PAD + it] = ri[0]; vals[ri[0]] = -1e30f; }
    __syncthreads();
  }
}

// ---------------------------------------------------------------------------
// Kernel C: column means of V per (b,h)
// ---------------------------------------------------------------------------
__global__ __launch_bounds__(256) void vmean_kernel(
    const float* __restrict__ V, float* __restrict__ vm) {
  __shared__ float red[256];
  int bh = blockIdx.x, t = threadIdx.x;
  int d = t & 63, part = t >> 6;
  const float* vb = V + (size_t)bh * LQ * DD;
  float s = 0.0f;
  for (int r = part; r < LQ; r += 4) s += vb[(size_t)r * DD + d];
  red[t] = s;
  __syncthreads();
  if (part == 0)
    vm[bh * DD + d] = (red[d] + red[64 + d] + red[128 + d] + red[192 + d]) * (1.0f / LQ);
}

// ---------------------------------------------------------------------------
// Kernel D: broadcast mean(V) into the whole output (overwritten for top-40)
// ---------------------------------------------------------------------------
__global__ __launch_bounds__(256) void fill_kernel(
    const float* __restrict__ vm, float* __restrict__ out) {
  size_t i = (size_t)blockIdx.x * 256 + threadIdx.x;
  int bh = (int)(i >> 17);          // / (LQ*DD)
  int d  = (int)(i & 63);
  out[i] = vm[bh * DD + d];
}

// ---------------------------------------------------------------------------
// Kernel E: flash-style attention for the top-40 queries of each (b,h).
// K/V chunks are double-buffered into LDS with GLOBAL_LOAD_ASYNC_TO_LDS_B128
// (ASYNCcnt-tracked, overlapped with compute); a cheap LDS->LDS convert pass
// produces the f16 tiles feeding v_wmma_f32_16x16x32_f16.
// ---------------------------------------------------------------------------
__global__ __launch_bounds__(128) void attn_kernel(
    const float* __restrict__ Q, const float* __restrict__ K,
    const float* __restrict__ V, const int* __restrict__ Mtop,
    float* __restrict__ out) {
  constexpr int CH     = 64;          // keys per chunk
  constexpr int CHF    = CH * DD;     // floats per chunk (4096 = 16KB)
  constexpr int NCHUNK = LQ / CH;     // 32
  constexpr int PADW   = 72;          // row pad to rotate LDS banks
  __shared__ float    Kraw[2][CHF];   // async landing zones (double buffer)
  __shared__ float    Vraw[2][CHF];
  __shared__ _Float16 Qs[4][16][PADW];
  __shared__ _Float16 Ks[CH][PADW];
  __shared__ _Float16 Vs[CH][PADW];
  __shared__ _Float16 Ps[4][16][PADW];

  int bh   = blockIdx.x;
  int w    = threadIdx.x >> 5;
  int lane = threadIdx.x & 31;
  const int*   mt = Mtop + bh * NTOP_PAD;
  const float* Qb = Q + (size_t)bh * LQ * DD;
  const float* Kb = K + (size_t)bh * LQ * DD;
  const float* Vb = V + (size_t)bh * LQ * DD;

  // stage this wave's 16 gathered query rows, f32 -> f16
  for (int rr = 0; rr < 16; ++rr) {
    const float* qp = Qb + (size_t)mt[w * 16 + rr] * DD;
    Qs[w][rr][lane]      = (_Float16)qp[lane];
    Qs[w][rr][lane + 32] = (_Float16)qp[lane + 32];
  }

  // fire the async copy of one K/V chunk into raw buffer `buf`
  // (8+8 wave-wide B128 instructions per wave -> ASYNCcnt += 16 per wave)
  auto issue_chunk = [&](int base, int buf) {
    const float* kg = Kb + (size_t)base * DD;
    const float* vg = Vb + (size_t)base * DD;
    unsigned kl = (unsigned)(uintptr_t)&Kraw[buf][0];
    unsigned vl = (unsigned)(uintptr_t)&Vraw[buf][0];
#pragma unroll
    for (int i = 0; i < 8; ++i) {
      int off = (i * 128 + (int)threadIdx.x) * 4;  // 16B granule per lane
      unsigned     l0 = kl + off * 4;
      const float* g0 = kg + off;
      asm volatile("global_load_async_to_lds_b128 %0, %1, off"
                   :: "v"(l0), "v"(g0) : "memory");
      unsigned     l1 = vl + off * 4;
      const float* g1 = vg + off;
      asm volatile("global_load_async_to_lds_b128 %0, %1, off"
                   :: "v"(l1), "v"(g1) : "memory");
    }
  };

  issue_chunk(0, 0);
  __syncthreads();

  int row16 = lane & 15;
  int kbase = (lane & 16) ? 8 : 0;   // A/B fragment K-offset for the lane half

  // Q A-fragments are chunk-invariant: build once (k0 = 0 and 32)
  v16h aQ[2];
  for (int kc = 0; kc < 2; ++kc)
    for (int e = 0; e < 16; ++e) {
      int kk = kc * 32 + e + (e < 8 ? 0 : 8) + kbase;
      aQ[kc][e] = Qs[w][row16][kk];
    }

  float m_i[8], l_i[8], O[4][8];
  for (int r = 0; r < 8; ++r) { m_i[r] = -1e30f; l_i[r] = 0.0f; }
  for (int t = 0; t < 4; ++t)
    for (int r = 0; r < 8; ++r) O[t][r] = 0.0f;

  const float scale = 0.125f;  // 1/sqrt(64)

  for (int c = 0; c < NCHUNK; ++c) {
    int buf = c & 1;
    if (c + 1 < NCHUNK) {
      issue_chunk((c + 1) * CH, buf ^ 1);           // prefetch next chunk
      asm volatile("s_wait_asynccnt 0x10" ::: "memory");  // chunk c landed
    } else {
      asm volatile("s_wait_asynccnt 0x0" ::: "memory");
    }
    __syncthreads();  // chunk c visible block-wide; prior reads of Ks/Vs done

    // LDS->LDS convert pass: raw f32 chunk -> f16 compute tiles
    {
      int r = threadIdx.x >> 1;
      int h = (threadIdx.x & 1) * 32;
#pragma unroll
      for (int i = 0; i < 32; ++i) {
        Ks[r][h + i] = (_Float16)Kraw[buf][r * DD + h + i];
        Vs[r][h + i] = (_Float16)Vraw[buf][r * DD + h + i];
      }
    }
    __syncthreads();

    // S = Q(16x64) @ K_chunk^T -> 4 C fragments (8 WMMA)
    float p[4][8];
    for (int f = 0; f < 4; ++f) {
      v8f cacc = {0.f, 0.f, 0.f, 0.f, 0.f, 0.f, 0.f, 0.f};
      for (int kc = 0; kc < 2; ++kc) {
        v16h b;
        for (int e = 0; e < 16; ++e) {
          int kk = kc * 32 + e + (e < 8 ? 0 : 8) + kbase;
          b[e] = Ks[f * 16 + row16][kk];
        }
        cacc = __builtin_amdgcn_wmma_f32_16x16x32_f16(false, aQ[kc], false, b,
                                                      (short)0, cacc, false, false);
      }
      for (int r = 0; r < 8; ++r) p[f][r] = cacc[r] * scale;
    }

    // online softmax: row r lives on one 16-lane half at reg r
    for (int r = 0; r < 8; ++r) {
      float mx = redmax16(fmaxf(fmaxf(p[0][r], p[1][r]), fmaxf(p[2][r], p[3][r])));
      float mnew  = fmaxf(m_i[r], mx);
      float alpha = __expf(m_i[r] - mnew);
      float rsum  = 0.0f;
      for (int f = 0; f < 4; ++f) {
        float e = __expf(p[f][r] - mnew);
        p[f][r] = e;
        rsum += e;
      }
      rsum   = redsum16(rsum);
      l_i[r] = l_i[r] * alpha + rsum;
      m_i[r] = mnew;
      for (int t = 0; t < 4; ++t) O[t][r] *= alpha;
    }

    // C-layout -> linear LDS (per-wave private region), f32 -> f16
    int prow = (lane & 16) ? 8 : 0;
    for (int f = 0; f < 4; ++f)
      for (int r = 0; r < 8; ++r)
        Ps[w][prow + r][f * 16 + row16] = (_Float16)p[f][r];
    __syncthreads();

    // O += P(16x64) @ V_chunk(64x64) -> 8 WMMA
    for (int kc = 0; kc < 2; ++kc) {
      v16h a;
      for (int e = 0; e < 16; ++e) {
        int kk = kc * 32 + e + (e < 8 ? 0 : 8) + kbase;
        a[e] = Ps[w][row16][kk];
      }
      for (int t = 0; t < 4; ++t) {
        v16h b;
        for (int e = 0; e < 16; ++e) {
          int kk = kc * 32 + e + (e < 8 ? 0 : 8) + kbase;
          b[e] = Vs[kk][t * 16 + row16];
        }
        v8f cacc;
        for (int r = 0; r < 8; ++r) cacc[r] = O[t][r];
        cacc = __builtin_amdgcn_wmma_f32_16x16x32_f16(false, a, false, b,
                                                      (short)0, cacc, false, false);
        for (int r = 0; r < 8; ++r) O[t][r] = cacc[r];
      }
    }
  }

  // normalize and scatter the 40 valid rows
  int prow = (lane & 16) ? 8 : 0;
  for (int r = 0; r < 8; ++r) {
    int slot = w * 16 + prow + r;
    if (slot < NTOP) {
      float  inv = 1.0f / l_i[r];
      float* op  = out + ((size_t)bh * LQ + mt[slot]) * DD;
      for (int t = 0; t < 4; ++t) op[t * 16 + row16] = O[t][r] * inv;
    }
  }
}

// ---------------------------------------------------------------------------
extern "C" void kernel_launch(void* const* d_in, const int* in_sizes, int n_in,
                              void* d_out, int out_size, void* d_ws, size_t ws_size,
                              hipStream_t stream) {
  const float* Q   = (const float*)d_in[0];
  const float* K   = (const float*)d_in[1];
  const float* V   = (const float*)d_in[2];
  const int*   idx = (const int*)d_in[3];
  float* out = (float*)d_out;

  float* Mbuf = (float*)d_ws;                         // B*H*L floats
  int*   Mtop = (int*)(Mbuf + BN * HN * LQ);          // B*H*64 ints
  float* vm   = (float*)(Mtop + BN * HN * NTOP_PAD);  // B*H*64 floats

  mscore_kernel<<<BN * HN * LQ / 8, 256, 0, stream>>>(Q, K, idx, Mbuf);
  topk_kernel  <<<BN * HN,          256, 0, stream>>>(Mbuf, Mtop);
  vmean_kernel <<<BN * HN,          256, 0, stream>>>(V, vm);
  fill_kernel  <<<(BN * HN * LQ * DD) / 256, 256, 0, stream>>>(vm, out);
  attn_kernel  <<<BN * HN,          128, 0, stream>>>(Q, K, V, Mtop, out);
}
